// SmallThinkerMoeBlock_42099269435440
// MI455X (gfx1250) — compile-verified, matrix operand order checked by reading, not run on hardware
//
#include <hip/hip_runtime.h>
#include <hip/hip_bf16.h>
#include <math.h>

#define T_TOK 2048
#define HDIM  1536
#define IDIM  768
#define NEXP  32
#define TOPK  4
#define BM    32

typedef _Float16 v16h __attribute__((ext_vector_type(16)));
typedef _Float16 v8h  __attribute__((ext_vector_type(8)));
typedef float    v8f  __attribute__((ext_vector_type(8)));

// ---------------------------------------------------------------------------
// WMMA helpers (CDNA5 gfx1250): D = A(16x32 f16) x B(32x16 f16) + C(16x16 f32)
// Per-lane K mapping (ISA 7.12.2, 16-bit operands):
//   lanes 0-15 : halves 0-7 -> K=kb+0..7,  halves 8-15 -> K=kb+16..23
//   lanes 16-31: halves 0-7 -> K=kb+8..15, halves 8-15 -> K=kb+24..31
// A: M = lane%16 (row).  B: N = lane%16 (row of W, since B = W^T column).
// ---------------------------------------------------------------------------
__device__ __forceinline__ v8f wmma_f16(v16h a, v16h b, v8f c) {
  return __builtin_amdgcn_wmma_f32_16x16x32_f16(false, a, false, b,
                                                (short)0, c, false, false);
}

// Fragment from an f16 row in LDS (two 16B ds_load_b128)
__device__ __forceinline__ v16h load_frag_lds(const _Float16* row, int kb, int klo) {
  v8h lo = *(const v8h*)(row + kb + klo);
  v8h hi = *(const v8h*)(row + kb + klo + 16);
  v16h r;
#pragma unroll
  for (int i = 0; i < 8; ++i) { r[i] = lo[i]; r[i + 8] = hi[i]; }
  return r;
}

// Fragment from an fp32 row in global memory, converted to f16 inline
__device__ __forceinline__ v16h load_frag_gf32(const float* __restrict__ row,
                                               int kb, int klo) {
  const float4 a = *(const float4*)(row + kb + klo);
  const float4 b = *(const float4*)(row + kb + klo + 4);
  const float4 c = *(const float4*)(row + kb + klo + 16);
  const float4 d = *(const float4*)(row + kb + klo + 20);
  v16h r;
  r[0]  = (_Float16)a.x; r[1]  = (_Float16)a.y; r[2]  = (_Float16)a.z; r[3]  = (_Float16)a.w;
  r[4]  = (_Float16)b.x; r[5]  = (_Float16)b.y; r[6]  = (_Float16)b.z; r[7]  = (_Float16)b.w;
  r[8]  = (_Float16)c.x; r[9]  = (_Float16)c.y; r[10] = (_Float16)c.z; r[11] = (_Float16)c.w;
  r[12] = (_Float16)d.x; r[13] = (_Float16)d.y; r[14] = (_Float16)d.z; r[15] = (_Float16)d.w;
  return r;
}

// ---------------------------------------------------------------------------
// Kernel 0: zero the output accumulator and per-expert counters
// ---------------------------------------------------------------------------
__global__ void moe_init_kernel(float* __restrict__ out, int* __restrict__ counts,
                                long n) {
  long i = (long)blockIdx.x * blockDim.x + threadIdx.x;
  long stride = (long)gridDim.x * blockDim.x;
  for (long j = i; j < n; j += stride) out[j] = 0.f;
  if (i < NEXP) counts[i] = 0;
}

// ---------------------------------------------------------------------------
// Kernel 1: router logits + top-4 + softmax + expert bucketing
// One wave32 per token; lane == expert (E == 32 exactly fills a wave).
// ---------------------------------------------------------------------------
__global__ void moe_router_kernel(const float* __restrict__ rin,
                                  const float* __restrict__ rw,
                                  int* __restrict__ counts,
                                  int* __restrict__ tok_ids,
                                  float* __restrict__ tok_gate) {
  const int lane = threadIdx.x & 31;
  const int wv   = threadIdx.x >> 5;
  const int t    = blockIdx.x * (blockDim.x >> 5) + wv;
  if (t >= T_TOK) return;

  const float4* x = (const float4*)(rin + (size_t)t * HDIM);
  const float4* w = (const float4*)(rw + (size_t)lane * HDIM);
  float acc = 0.f;
  for (int i = 0; i < HDIM / 4; ++i) {
    float4 xv = x[i], wv4 = w[i];
    acc += xv.x * wv4.x + xv.y * wv4.y + xv.z * wv4.z + xv.w * wv4.w;
  }

  // top-4 via repeated wave32 argmax (ties -> lower index, like lax.top_k)
  float cur = acc;
  float vals[TOPK];
  int   idxs[TOPK];
#pragma unroll
  for (int r = 0; r < TOPK; ++r) {
    float m = cur; int mi = lane;
#pragma unroll
    for (int off = 16; off >= 1; off >>= 1) {
      float om = __shfl_xor(m, off, 32);
      int   oi = __shfl_xor(mi, off, 32);
      if (om > m || (om == m && oi < mi)) { m = om; mi = oi; }
    }
    vals[r] = m; idxs[r] = mi;
    if (lane == mi) cur = -__builtin_inff();
  }

  // softmax over the 4 selected logits (vals[0] is the max)
  float s = 0.f, ev[TOPK];
#pragma unroll
  for (int r = 0; r < TOPK; ++r) { ev[r] = __expf(vals[r] - vals[0]); s += ev[r]; }

  if (lane < TOPK) {
    const int ex = idxs[lane];
    const float g = ev[lane] / s;
    const int pos = atomicAdd(&counts[ex], 1);
    tok_ids[ex * T_TOK + pos]  = t;
    tok_gate[ex * T_TOK + pos] = g;
  }
}

// ---------------------------------------------------------------------------
// Kernel 2: fused expert MLP per (expert, 32-token tile)
//   GEMM1: act = relu(X W_g^T) * (X W_u^T)   [32 x 768]
//   GEMM2: out += gate * (act W_d^T)         [32 x 1536], atomic scatter
// 256 threads = 8 waves. Each wave computes BOTH 16-row M-halves for its
// N-tiles so every B fragment (global fp32 -> f16 cvt) is loaded once and
// feeds two v_wmma issues -> halves weight fetch + conversion VALU work.
// LDS: Xs f16 [32][1536] (96KB) + As f16 [32][768] (48KB) + toks/gates.
// 144.3KB -> 2 workgroups / 320KB WGP.
// ---------------------------------------------------------------------------
__global__ void moe_expert_kernel(const float* __restrict__ hs,
                                  const float* __restrict__ w13,
                                  const float* __restrict__ w2,
                                  const int* __restrict__ counts,
                                  const int* __restrict__ tok_ids,
                                  const float* __restrict__ tok_gate,
                                  float* __restrict__ out) {
  const int e    = blockIdx.y;
  const int cnt  = counts[e];
  const int row0 = blockIdx.x * BM;
  if (row0 >= cnt) return;

  extern __shared__ char smem_raw[];
  _Float16* Xs   = (_Float16*)smem_raw;        // [BM][HDIM]
  _Float16* As   = Xs + BM * HDIM;             // [BM][IDIM]
  int*      toks = (int*)(As + BM * IDIM);     // [BM]
  float*    gts  = (float*)(toks + BM);        // [BM]

  const int tid  = threadIdx.x;
  const int lane = tid & 31;
  const int wv   = tid >> 5;
  const int col  = lane & 15;                  // N (or M for A) = lane%16
  const int klo  = (lane < 16) ? 0 : 8;        // K sub-offset per lane half
  const int mhi  = (lane < 16) ? 0 : 8;        // D-matrix row offset per lane half

  if (tid < BM) {
    const int j = row0 + tid;
    if (j < cnt) { toks[tid] = tok_ids[e * T_TOK + j]; gts[tid] = tok_gate[e * T_TOK + j]; }
    else         { toks[tid] = -1;                     gts[tid] = 0.f; }
  }
  __syncthreads();

  // Gather 32 token rows -> f16 LDS tile (coalesced float4 loads)
  for (int idx = tid; idx < BM * (HDIM / 4); idx += blockDim.x) {
    const int r = idx / (HDIM / 4);
    const int c = (idx % (HDIM / 4)) * 4;
    const int tk = toks[r];
    float4 v = make_float4(0.f, 0.f, 0.f, 0.f);
    if (tk >= 0) v = *(const float4*)(hs + (size_t)tk * HDIM + c);
    _Float16* dst = Xs + r * HDIM + c;
    dst[0] = (_Float16)v.x; dst[1] = (_Float16)v.y;
    dst[2] = (_Float16)v.z; dst[3] = (_Float16)v.w;
  }
  __syncthreads();

  // A rows for this lane, both M-halves
  const _Float16* Arow0 = Xs + col * HDIM;          // M-half 0 (rows 0..15)
  const _Float16* Arow1 = Xs + (16 + col) * HDIM;   // M-half 1 (rows 16..31)
  const float* w13e = w13 + (size_t)e * (2 * IDIM) * HDIM;

  // ---- GEMM1 + relu-gate, write act tile to LDS as f16 ----
  // 48 gate tiles / 8 waves = 6 per wave; 4 WMMAs per K-step sharing bg/bu.
  for (int gt = wv; gt < IDIM / 16; gt += 8) {
    v8f aG0 = {0.f,0.f,0.f,0.f,0.f,0.f,0.f,0.f};
    v8f aG1 = aG0, aU0 = aG0, aU1 = aG0;
    const float* Bg = w13e + (size_t)(gt * 16 + col) * HDIM;          // gate rows
    const float* Bu = w13e + (size_t)(IDIM + gt * 16 + col) * HDIM;   // up rows
    for (int kb = 0; kb < HDIM; kb += 32) {
      __builtin_prefetch(Bg + kb + 32, 0, 1);   // global_prefetch next chunk
      __builtin_prefetch(Bu + kb + 32, 0, 1);
      v16h bg = load_frag_gf32(Bg, kb, klo);
      v16h bu = load_frag_gf32(Bu, kb, klo);
      v16h a0 = load_frag_lds(Arow0, kb, klo);
      v16h a1 = load_frag_lds(Arow1, kb, klo);
      aG0 = wmma_f16(a0, bg, aG0);
      aG1 = wmma_f16(a1, bg, aG1);
      aU0 = wmma_f16(a0, bu, aU0);
      aU1 = wmma_f16(a1, bu, aU1);
    }
#pragma unroll
    for (int r8 = 0; r8 < 8; ++r8) {
      const float g0 = aG0[r8];
      const float g1 = aG1[r8];
      const float act0 = (g0 > 0.f ? g0 : 0.f) * aU0[r8];
      const float act1 = (g1 > 0.f ? g1 : 0.f) * aU1[r8];
      As[(mhi + r8) * IDIM + gt * 16 + col]      = (_Float16)act0;
      As[(16 + mhi + r8) * IDIM + gt * 16 + col] = (_Float16)act1;
    }
  }
  __syncthreads();

  // ---- GEMM2: down-proj, scale by gate, atomic scatter into d_out ----
  // 96 N-tiles / 8 waves = 12 per wave; 2 WMMAs per K-step sharing b.
  const float* w2e = w2 + (size_t)e * HDIM * IDIM;
  const _Float16* Crow0 = As + col * IDIM;
  const _Float16* Crow1 = As + (16 + col) * IDIM;
  for (int nt = wv; nt < HDIM / 16; nt += 8) {
    v8f acc0 = {0.f,0.f,0.f,0.f,0.f,0.f,0.f,0.f};
    v8f acc1 = acc0;
    const float* B = w2e + (size_t)(nt * 16 + col) * IDIM;
    for (int kb = 0; kb < IDIM; kb += 32) {
      __builtin_prefetch(B + kb + 32, 0, 1);
      v16h b  = load_frag_gf32(B, kb, klo);
      v16h a0 = load_frag_lds(Crow0, kb, klo);
      v16h a1 = load_frag_lds(Crow1, kb, klo);
      acc0 = wmma_f16(a0, b, acc0);
      acc1 = wmma_f16(a1, b, acc1);
    }
#pragma unroll
    for (int r8 = 0; r8 < 8; ++r8) {
      const int m0 = mhi + r8;
      const int m1 = 16 + mhi + r8;
      const int tk0 = toks[m0];
      const int tk1 = toks[m1];
      if (tk0 >= 0)
        atomicAdd(out + (size_t)tk0 * HDIM + nt * 16 + col, gts[m0] * acc0[r8]);
      if (tk1 >= 0)
        atomicAdd(out + (size_t)tk1 * HDIM + nt * 16 + col, gts[m1] * acc1[r8]);
    }
  }
}

// ---------------------------------------------------------------------------
extern "C" void kernel_launch(void* const* d_in, const int* in_sizes, int n_in,
                              void* d_out, int out_size, void* d_ws, size_t ws_size,
                              hipStream_t stream) {
  const float* rin = (const float*)d_in[0];   // router_input [T,H]
  const float* hs  = (const float*)d_in[1];   // hidden_states [T,H]
  const float* rw  = (const float*)d_in[2];   // router_w [E,H]
  const float* w13 = (const float*)d_in[3];   // [E, 2I, H]
  const float* w2  = (const float*)d_in[4];   // [E, H, I]
  float* out = (float*)d_out;

  // workspace layout: counts[E] | tok_ids[E*T] | tok_gate[E*T]
  char* ws = (char*)d_ws;
  int*   counts   = (int*)ws;
  int*   tok_ids  = (int*)(ws + 256);
  float* tok_gate = (float*)(ws + 256 + (size_t)NEXP * T_TOK * sizeof(int));

  moe_init_kernel<<<512, 256, 0, stream>>>(out, counts, (long)T_TOK * HDIM);
  moe_router_kernel<<<T_TOK / 8, 256, 0, stream>>>(rin, rw, counts, tok_ids, tok_gate);

  const size_t smem = (size_t)(BM * HDIM + BM * IDIM) * sizeof(_Float16)
                    + BM * sizeof(int) + BM * sizeof(float);   // ~144.3 KB < 320 KB/WGP
  (void)hipFuncSetAttribute(reinterpret_cast<const void*>(moe_expert_kernel),
                            hipFuncAttributeMaxDynamicSharedMemorySize, (int)smem);
  dim3 grid(T_TOK / BM, NEXP);
  moe_expert_kernel<<<grid, 256, smem, stream>>>(hs, w13, w2, counts, tok_ids,
                                                 tok_gate, out);
}
